// DemoEnhancer_66460323938504
// MI455X (gfx1250) — compile-verified, matrix operand order checked by reading, not run on hardware
//
#include <hip/hip_runtime.h>

#define LUT_D 33
#define NPTS (LUT_D * LUT_D * LUT_D)   // 35937
#define HID 256
#define HPAD 264                       // 256 + 8 halves pad -> conflict-free ds_load_b128

typedef __attribute__((ext_vector_type(16))) _Float16 v16h;
typedef __attribute__((ext_vector_type(8)))  _Float16 v8h;
typedef __attribute__((ext_vector_type(8)))  float    v8f;
typedef __attribute__((ext_vector_type(4)))  float    f4;   // vector float4 (NT-load legal)

// Branch-free tanh: 1 - 2/(exp(2x)+1); exact saturation at +/-1, uses v_exp_f32.
__device__ __forceinline__ float fast_tanh(float x) {
    float e = __expf(2.0f * x);
    return 1.0f - 2.0f / (e + 1.0f);
}

// ---------------------------------------------------------------------------
// Kernel 1: convert W1..W3 (fp32, (256,256) row-major, x@W sums over dim0=K)
// into f16 fragments in CDNA5 WMMA 16-bit B-matrix order.
// Fragment flat index within a layer: ((ntile*8 + kstep)*32 + lane)*16 + h
//   n      = ntile*16 + (lane & 15)
//   kbase  = (lane >> 4) * 8
//   v=h>>1, sub=h&1
//   klocal = v<4 ? kbase + 2v + sub : 16 + kbase + 2(v-4) + sub
//   k      = kstep*32 + klocal
// ---------------------------------------------------------------------------
__global__ void nilut_prep_weights(const float* __restrict__ W1,
                                   const float* __restrict__ W2,
                                   const float* __restrict__ W3,
                                   _Float16* __restrict__ Wsw) {
    const int per_layer = 16 * 8 * 32 * 16;  // 65536
    int t = blockIdx.x * blockDim.x + threadIdx.x;
    if (t >= 3 * per_layer) return;
    int L  = t / per_layer;
    int r  = t % per_layer;
    int nt = r >> 12;            // / 4096
    int r2 = r & 4095;
    int ks = r2 >> 9;            // / 512
    int r3 = r2 & 511;
    int lane = r3 >> 4;
    int h    = r3 & 15;
    int n     = nt * 16 + (lane & 15);
    int kbase = (lane >> 4) * 8;
    int v = h >> 1, sub = h & 1;
    int klocal = (v < 4) ? (kbase + 2 * v + sub) : (16 + kbase + 2 * (v - 4) + sub);
    int k = ks * 32 + klocal;
    const float* W = (L == 0) ? W1 : (L == 1) ? W2 : W3;
    Wsw[t] = (_Float16)W[k * HID + n];
}

// ---------------------------------------------------------------------------
// Kernel 2: build LUT. One block = 16 LUT vertices, 256 threads = 8 waves.
// Two MLP passes (full params, zeroed params); lut = net(x) - net(x0) + rgb.
// ---------------------------------------------------------------------------
__global__ __launch_bounds__(256)
void nilut_lut_kernel(const float* __restrict__ param,
                      const float* __restrict__ W0, const float* __restrict__ b0,
                      const float* __restrict__ b1, const float* __restrict__ b2,
                      const float* __restrict__ b3,
                      const float* __restrict__ W4, const float* __restrict__ b4,
                      const _Float16* __restrict__ Wsw,
                      float* __restrict__ lut) {     // RGBA packed, 4 floats/vertex
    __shared__ __attribute__((aligned(16))) _Float16 Hb[2][16][HPAD];
    __shared__ float Xf[16][12];
    __shared__ float rgbS[16][3];
    __shared__ float out0[16][3];
    __shared__ float prm[5];

    const int tid  = threadIdx.x;
    const int wave = tid >> 5;
    const int lane = tid & 31;
    const int mrow = lane & 15;
    const int kbase = (lane >> 4) * 8;
    const int blk = blockIdx.x;

    if (tid < 5) prm[tid] = param[tid];

    if (tid < 16) {
        int n = blk * 16 + tid;
        if (n >= NPTS) n = NPTS - 1;
        int kk = n % LUT_D;
        int jj = (n / LUT_D) % LUT_D;
        int ii = n / (LUT_D * LUT_D);
        float c0 = kk * (1.0f / (LUT_D - 1));   // channel 0 (innermost)
        float c1 = jj * (1.0f / (LUT_D - 1));
        float c2 = ii * (1.0f / (LUT_D - 1));
        rgbS[tid][0] = c0; rgbS[tid][1] = c1; rgbS[tid][2] = c2;
        float lo  = fminf(fminf(c0, c1), c2);
        float hi  = fmaxf(fmaxf(c0, c1), c2);
        float mid = c0 + c1 + c2 - lo - hi;
        Xf[tid][0] = c0; Xf[tid][1] = c1; Xf[tid][2] = c2;
        Xf[tid][8] = lo; Xf[tid][9] = mid; Xf[tid][10] = hi;
        Xf[tid][11] = 0.0f;
    }
    __syncthreads();

    for (int pass = 0; pass < 2; ++pass) {
        if (tid < 16) {
            #pragma unroll
            for (int p = 0; p < 5; ++p) Xf[tid][3 + p] = (pass == 0) ? prm[p] : 0.0f;
        }
        __syncthreads();

        // ---- Layer 0: (16 x 11) @ (11 x 256) + b0, relu ----  (VALU, tiny K)
        {
            float w[11];
            #pragma unroll
            for (int k = 0; k < 11; ++k) w[k] = W0[k * HID + tid];
            float bb = b0[tid];
            #pragma unroll
            for (int m = 0; m < 16; ++m) {
                float acc = bb;
                #pragma unroll
                for (int k = 0; k < 11; ++k) acc += Xf[m][k] * w[k];
                Hb[0][m][tid] = (_Float16)fmaxf(acc, 0.0f);
            }
        }
        __syncthreads();

        // ---- Layers 1..3: (16 x 256) @ (256 x 256) + b, tanh  via WMMA ----
        int cur = 0;
        for (int L = 0; L < 3; ++L) {
            const _Float16* Wl = Wsw + L * 65536;
            const float* bias = (L == 0) ? b1 : (L == 1) ? b2 : b3;
            const int nt0 = wave * 2, nt1 = wave * 2 + 1;
            v8f acc0 = {}; v8f acc1 = {};
            #pragma unroll
            for (int ks = 0; ks < 8; ++ks) {
                int col = ks * 32 + kbase;
                v8h alo = *(const v8h*)&Hb[cur][mrow][col];
                v8h ahi = *(const v8h*)&Hb[cur][mrow][col + 16];
                v16h a = __builtin_shufflevector(alo, ahi,
                        0, 1, 2, 3, 4, 5, 6, 7, 8, 9, 10, 11, 12, 13, 14, 15);
                v16h bf0 = *(const v16h*)(Wl + (((nt0 * 8 + ks) * 32 + lane) << 4));
                v16h bf1 = *(const v16h*)(Wl + (((nt1 * 8 + ks) * 32 + lane) << 4));
                acc0 = __builtin_amdgcn_wmma_f32_16x16x32_f16(
                        false, a, false, bf0, (short)0, acc0, false, false);
                acc1 = __builtin_amdgcn_wmma_f32_16x16x32_f16(
                        false, a, false, bf1, (short)0, acc1, false, false);
            }
            int nxt = cur ^ 1;
            int nc0 = nt0 * 16 + (lane & 15);
            int nc1 = nt1 * 16 + (lane & 15);
            int mb  = (lane >> 4) * 8;
            float bia0 = bias[nc0], bia1 = bias[nc1];
            #pragma unroll
            for (int i = 0; i < 8; ++i) {
                Hb[nxt][mb + i][nc0] = (_Float16)fast_tanh(acc0[i] + bia0);
                Hb[nxt][mb + i][nc1] = (_Float16)fast_tanh(acc1[i] + bia1);
            }
            cur = nxt;
            __syncthreads();
        }

        // ---- Layer 4: (16 x 256) @ (256 x 3) + b4 ----  (VALU, tiny N)
        if (tid < 48) {
            int m = tid / 3, c = tid % 3;
            float acc = b4[c];
            for (int k = 0; k < HID; ++k)
                acc += (float)Hb[cur][m][k] * W4[k * 3 + c];
            if (pass == 0) {
                out0[m][c] = acc;
            } else {
                int n = blk * 16 + m;
                if (n < NPTS) {
                    float val = out0[m][c] - acc + rgbS[m][c];
                    lut[n * 4 + c] = val;
                }
            }
        }
        __syncthreads();
    }
    if (tid < 16) {
        int n = blk * 16 + tid;
        if (n < NPTS) lut[n * 4 + 3] = 0.0f;
    }
}

// ---------------------------------------------------------------------------
// Kernel 3: trilinear LUT apply, planar image, vectorized, NT-hinted stream.
// lut is RGBA-packed, index [b][g][r] (r innermost) -> one f4 load per corner.
// ---------------------------------------------------------------------------
__device__ __forceinline__ f4 lut_one(float r, float g, float b,
                                      const f4* __restrict__ lut) {
    float sr = r * (float)(LUT_D - 1);
    float sg = g * (float)(LUT_D - 1);
    float sb = b * (float)(LUT_D - 1);
    int ir = (int)floorf(sr); ir = ir < 0 ? 0 : (ir > LUT_D - 2 ? LUT_D - 2 : ir);
    int ig = (int)floorf(sg); ig = ig < 0 ? 0 : (ig > LUT_D - 2 ? LUT_D - 2 : ig);
    int ib = (int)floorf(sb); ib = ib < 0 ? 0 : (ib > LUT_D - 2 ? LUT_D - 2 : ib);
    float fr = sr - (float)ir, fg = sg - (float)ig, fb = sb - (float)ib;
    int base = (ib * LUT_D + ig) * LUT_D + ir;
    const int SB = LUT_D * LUT_D;  // 1089
    f4 c000 = lut[base],              c001 = lut[base + 1];
    f4 c010 = lut[base + LUT_D],      c011 = lut[base + LUT_D + 1];
    f4 c100 = lut[base + SB],         c101 = lut[base + SB + 1];
    f4 c110 = lut[base + SB + LUT_D], c111 = lut[base + SB + LUT_D + 1];
    float w000 = (1 - fb) * (1 - fg) * (1 - fr);
    float w001 = (1 - fb) * (1 - fg) * fr;
    float w010 = (1 - fb) * fg * (1 - fr);
    float w011 = (1 - fb) * fg * fr;
    float w100 = fb * (1 - fg) * (1 - fr);
    float w101 = fb * (1 - fg) * fr;
    float w110 = fb * fg * (1 - fr);
    float w111 = fb * fg * fr;
    return w000 * c000 + w001 * c001 + w010 * c010 + w011 * c011
         + w100 * c100 + w101 * c101 + w110 * c110 + w111 * c111;
}

__global__ __launch_bounds__(256)
void nilut_apply_kernel(const float* __restrict__ img,
                        const f4* __restrict__ lut,
                        float* __restrict__ out) {
    const int HW = 2160 * 3840;
    const int NQ = HW / 4;
    const f4* R  = (const f4*)img;
    const f4* G  = (const f4*)(img + HW);
    const f4* Bp = (const f4*)(img + 2 * HW);
    f4* Or = (f4*)out;
    f4* Og = (f4*)(out + HW);
    f4* Ob = (f4*)(out + 2 * HW);
    for (int q = blockIdx.x * blockDim.x + threadIdx.x; q < NQ;
         q += gridDim.x * blockDim.x) {
        // Non-temporal: image/output are streamed exactly once; keep LUT hot.
        f4 r4 = __builtin_nontemporal_load(&R[q]);
        f4 g4 = __builtin_nontemporal_load(&G[q]);
        f4 b4 = __builtin_nontemporal_load(&Bp[q]);
        f4 or4, og4, ob4;
        #pragma unroll
        for (int e = 0; e < 4; ++e) {
            f4 res = lut_one(r4[e], g4[e], b4[e], lut);
            or4[e] = res[0];
            og4[e] = res[1];
            ob4[e] = res[2];
        }
        __builtin_nontemporal_store(or4, &Or[q]);
        __builtin_nontemporal_store(og4, &Og[q]);
        __builtin_nontemporal_store(ob4, &Ob[q]);
    }
}

// ---------------------------------------------------------------------------
extern "C" void kernel_launch(void* const* d_in, const int* in_sizes, int n_in,
                              void* d_out, int out_size, void* d_ws, size_t ws_size,
                              hipStream_t stream) {
    (void)in_sizes; (void)n_in; (void)out_size; (void)ws_size;
    const float* img   = (const float*)d_in[0];
    const float* param = (const float*)d_in[1];
    const float* W0 = (const float*)d_in[2];
    const float* b0 = (const float*)d_in[3];
    const float* W1 = (const float*)d_in[4];
    const float* b1 = (const float*)d_in[5];
    const float* W2 = (const float*)d_in[6];
    const float* b2 = (const float*)d_in[7];
    const float* W3 = (const float*)d_in[8];
    const float* b3 = (const float*)d_in[9];
    const float* W4 = (const float*)d_in[10];
    const float* b4 = (const float*)d_in[11];

    _Float16* Wsw = (_Float16*)d_ws;                              // 384 KB
    float*    lut = (float*)((char*)d_ws + 3 * 65536 * sizeof(_Float16));

    const int prep_n = 3 * 65536;
    nilut_prep_weights<<<(prep_n + 255) / 256, 256, 0, stream>>>(W1, W2, W3, Wsw);

    const int nblocks = (NPTS + 15) / 16;                          // 2247
    nilut_lut_kernel<<<nblocks, 256, 0, stream>>>(param, W0, b0, b1, b2, b3,
                                                  W4, b4, Wsw, lut);

    const int HW = 2160 * 3840;
    nilut_apply_kernel<<<(HW / 4 + 255) / 256, 256, 0, stream>>>(img, (const f4*)lut,
                                                                 (float*)d_out);
}
